// NN_Model_19748259627524
// MI455X (gfx1250) — matrix operand order repaired and checked by reading, not compiled, not run on hardware
//
#include <hip/hip_runtime.h>
#include <hip/hip_bf16.h>

typedef __attribute__((ext_vector_type(16))) _Float16 v16h;
typedef __attribute__((ext_vector_type(8)))  float    v8f;

#define NROWS   16384
#define DDIM    17      // DATA_DIM+1
#define DH      64
#define LEN_MC  1088
#define M_PRIC  65
#define M0C     100
#define M1C     100
#define KP1     32      // K of GEMM1 padded 17 -> 32
#define CP      112     // c padded 100 -> 112
#define CT      7       // c tiles of 16

// workspace layout
#define WS_W0H_ELEMS (M0C*DH*KP1)                 // _Float16 [m][h][k]
#define WS_W1H_OFF   (WS_W0H_ELEMS*2)             // _Float16 [c][k]
#define WS_W1H_ELEMS (CP*DH)
#define WS_BIAS_OFF  (WS_W1H_OFF + WS_W1H_ELEMS*2) // float [CP]
#define OUT_TAIL     ((size_t)NROWS*M0C*M1C)
#define TAIL_ELEMS   (2*LEN_MC + 2*M_PRIC)

// Deterministic counter-based Gaussian (splitmix64 hash + Box-Muller).
// (JAX threefry noise cannot be matched bit-exactly; noise is fixed-key in the
// reference, so this is deterministic per call as required.)
__device__ __forceinline__ float nrand(unsigned stream, unsigned idx) {
    unsigned long long z = (((unsigned long long)stream << 32) | idx) + 0x9E3779B97F4A7C15ull;
    z = (z ^ (z >> 30)) * 0xBF58476D1CE4E5B9ull;
    z = (z ^ (z >> 27)) * 0x94D049BB133111EBull;
    z ^= z >> 31;
    unsigned u0 = (unsigned)z, u1 = (unsigned)(z >> 32);
    float u = ((u0 >> 9) + 0.5f) * (1.0f / 8388608.0f);   // (0,1)
    float r = sqrtf(-2.0f * logf(u));
    float th = 6.28318530717958647f * (u1 * 2.3283064365386963e-10f);
    return r * cosf(th);
}

// Sample weights into f16 WMMA-padded workspace + write output tail.
__global__ void sample_weights_kernel(const float* __restrict__ ms_vs,
                                      _Float16* __restrict__ w0h,
                                      _Float16* __restrict__ w1h,
                                      float* __restrict__ bias,
                                      float* __restrict__ out_tail) {
    int i = blockIdx.x * blockDim.x + threadIdx.x;
    if (i < WS_W0H_ELEMS) {                      // W0h[m][h][k]
        int m = i / (DH * KP1);
        int rem = i % (DH * KP1);
        int h = rem / KP1, k = rem % KP1;
        float v = 0.0f;
        if (k < DDIM) {
            int row = k * DH + h;                // samps_w0 row index
            float mw = ms_vs[row];
            float vw = fabsf(ms_vs[LEN_MC + M_PRIC + row]) + 1e-6f;
            v = mw + nrand(0u, (unsigned)(row * M0C + m)) * sqrtf(vw);
        }
        w0h[i] = (_Float16)v;
        return;
    }
    int j = i - WS_W0H_ELEMS;
    if (j < WS_W1H_ELEMS) {                      // W1h[c][k], k = samps_w1 row-1
        int c = j / DH, k = j % DH;
        float v = 0.0f;
        if (c < M1C) {
            int row = k + 1;
            float mp = ms_vs[LEN_MC + row];
            float vp = fabsf(ms_vs[2 * LEN_MC + M_PRIC + row]) + 1e-6f;
            v = mp + nrand(1u, (unsigned)(row * M1C + c)) * sqrtf(vp);
        }
        w1h[j] = (_Float16)v;
        return;
    }
    int b = j - WS_W1H_ELEMS;
    if (b < CP) {                                // bias = samps_w1 row 0
        float v = 0.0f;
        if (b < M1C) {
            float mp = ms_vs[LEN_MC];
            float vp = fabsf(ms_vs[2 * LEN_MC + M_PRIC]) + 1e-6f;
            v = mp + nrand(1u, (unsigned)b) * sqrtf(vp);
        }
        bias[b] = v;
        return;
    }
    int t = b - CP;                              // output tail: m_w0,v_w0,m_pri,v_pri
    if (t < TAIL_ELEMS) {
        float v;
        if (t < LEN_MC)                 v = ms_vs[t];
        else if (t < 2 * LEN_MC)        v = fabsf(ms_vs[LEN_MC + M_PRIC + (t - LEN_MC)]) + 1e-6f;
        else if (t < 2 * LEN_MC + M_PRIC) v = ms_vs[LEN_MC + (t - 2 * LEN_MC)];
        else                            v = fabsf(ms_vs[2 * LEN_MC + M_PRIC + (t - 2 * LEN_MC - M_PRIC)]) + 1e-6f;
        out_tail[t] = v;
    }
}

// Fused: Z = relu(X @ W0[:,:,m]); pred[n,m,:] = Z_n @ W1[1:,:] + W1[0,:]
// One block = one 16-row n-tile, 4 waves each owning m = wave + 4k.
__global__ __launch_bounds__(128) void mlp_sample_kernel(
    const float* __restrict__ x, const _Float16* __restrict__ w0h,
    const unsigned* __restrict__ w1h_u32, const float* __restrict__ bias,
    float* __restrict__ out)
{
    __shared__ __align__(32) _Float16 sX[16 * KP1];     // x tile, f16, K-padded
    __shared__ __align__(32) _Float16 sW1[CP * DH];     // [c][k]
    __shared__ float sBias[CP];
    __shared__ __align__(32) _Float16 sZ[4][16 * DH];   // per-wave relu buffer
    __shared__ float sOut[4][16 * CP];                  // per-wave store staging

    const int tid = threadIdx.x;
    const int n0 = blockIdx.x * 16;

    for (int i = tid; i < 16 * KP1; i += 128) {
        int r = i >> 5, k = i & 31;
        float v = (k < DDIM) ? x[(size_t)(n0 + r) * DDIM + k] : 0.0f;
        sX[i] = (_Float16)v;
    }
    unsigned* sW1u = (unsigned*)sW1;
    for (int i = tid; i < CP * DH / 2; i += 128) sW1u[i] = w1h_u32[i];
    for (int i = tid; i < CP; i += 128) sBias[i] = bias[i];
    __syncthreads();

    const int wave = tid >> 5;
    const int lane = tid & 31;
    const int lrow = lane & 15;
    const int lhi  = lane >> 4;

    // A1 operand (16x32 f16): lane holds M=lane%16, K = lhi*16 + s
    v16h a1;
#pragma unroll
    for (int s = 0; s < 16; ++s) {
        int K = lhi * 16 + s;
        a1[s] = sX[lrow * KP1 + K];
    }
    float biasr[CT];
#pragma unroll
    for (int t = 0; t < CT; ++t) biasr[t] = sBias[t * 16 + lrow];

    _Float16* zbuf = sZ[wave];
    float* obuf = sOut[wave];

    for (int m = wave; m < M0C; m += 4) {
        const _Float16* w0m = w0h + (size_t)m * (DH * KP1);

        // GEMM1: 16x64x32, 4 WMMA
        v8f acc[4];
#pragma unroll
        for (int t = 0; t < 4; ++t) {
            // B (32x16): lane N=lane%16, K=lhi*16 + s (contiguous 32B)
            v16h b1 = *(const v16h*)(w0m + (t * 16 + lrow) * KP1 + lhi * 16);
            v8f cz = {};
            acc[t] = __builtin_amdgcn_wmma_f32_16x16x32_f16(
                false, a1, false, b1, (short)0, cz, false, false);
        }
        // relu, D-layout -> row-major z in LDS
#pragma unroll
        for (int t = 0; t < 4; ++t)
#pragma unroll
            for (int r = 0; r < 8; ++r) {
                float v = acc[t][r];
                zbuf[(r + 8 * lhi) * DH + t * 16 + lrow] = (_Float16)(v > 0.0f ? v : 0.0f);
            }
        asm volatile("s_wait_dscnt 0" ::: "memory");

        // A2 operands (two K=32 slices of 16x64 z)
        v16h a2[2];
#pragma unroll
        for (int ks = 0; ks < 2; ++ks)
#pragma unroll
            for (int s = 0; s < 16; ++s) {
                int K = ks * 32 + lhi * 16 + s;
                a2[ks][s] = zbuf[lrow * DH + K];
            }

        // GEMM2: 16x112x64, 14 WMMA, bias in accumulator
#pragma unroll
        for (int ct = 0; ct < CT; ++ct) {
            v8f c;
#pragma unroll
            for (int r = 0; r < 8; ++r) c[r] = biasr[ct];
#pragma unroll
            for (int ks = 0; ks < 2; ++ks) {
                v16h b2 = *(const v16h*)(&sW1[(ct * 16 + lrow) * DH + ks * 32 + lhi * 16]);
                c = __builtin_amdgcn_wmma_f32_16x16x32_f16(
                    false, a2[ks], false, b2, (short)0, c, false, false);
            }
#pragma unroll
            for (int r = 0; r < 8; ++r)
                obuf[(r + 8 * lhi) * CP + ct * 16 + lrow] = c[r];
        }
        asm volatile("s_wait_dscnt 0" ::: "memory");

        // coalesced row stores: 16 rows x 100 floats (400B each)
#pragma unroll 4
        for (int r = 0; r < 16; ++r) {
            size_t base = (size_t)(n0 + r) * (M0C * M1C) + (size_t)m * M1C;
            for (int ci = lane; ci < M1C; ci += 32)
                out[base + ci] = obuf[r * CP + ci];
        }
    }
}

extern "C" void kernel_launch(void* const* d_in, const int* in_sizes, int n_in,
                              void* d_out, int out_size, void* d_ws, size_t ws_size,
                              hipStream_t stream) {
    const float* x     = (const float*)d_in[0];
    const float* ms_vs = (const float*)d_in[1];
    float* out = (float*)d_out;

    _Float16* w0h = (_Float16*)d_ws;
    _Float16* w1h = (_Float16*)((char*)d_ws + WS_W1H_OFF);
    float*    bia = (float*)((char*)d_ws + WS_BIAS_OFF);

    const int setup_elems = WS_W0H_ELEMS + WS_W1H_ELEMS + CP + TAIL_ELEMS;
    sample_weights_kernel<<<(setup_elems + 255) / 256, 256, 0, stream>>>(
        ms_vs, w0h, w1h, bia, out + OUT_TAIL);

    mlp_sample_kernel<<<NROWS / 16, 128, 0, stream>>>(
        x, w0h, (const unsigned*)w1h, bia, out);
}
